// SimpleTransformer_71124658422401
// MI455X (gfx1250) — matrix operand order repaired and checked by reading, not compiled, not run on hardware
//
#include <hip/hip_runtime.h>
#include <stdint.h>

// ---------------- model constants ----------------
#define NBATCH 64
#define NPAT   196          // valid tokens per image
#define TP     256          // padded tokens per image
#define PDIM   768
#define BDIM   1024
#define HEADS  8
#define VD     128
#define NDEPTH 6
#define NCLS   1000
#define EPSLN  1e-5f
#define ROWS   (NBATCH * NPAT)   // 12544 valid rows
#define PROWS  (NBATCH * TP)     // 16384 padded rows

typedef __bf16 v16bf __attribute__((ext_vector_type(16)));
typedef float  v8f   __attribute__((ext_vector_type(8)));

// Async global->LDS staging (gfx1250): use if this toolchain exposes the builtins.
#if defined(__AMDGCN__) && \
    __has_builtin(__builtin_amdgcn_global_load_async_to_lds_b128) && \
    __has_builtin(__builtin_amdgcn_s_wait_asynccnt)
#define HAVE_ASYNC 1
#else
#define HAVE_ASYNC 0
#endif

#if HAVE_ASYNC
typedef int v4i_ __attribute__((vector_size(4 * sizeof(int))));
#define GAS(p) ((__attribute__((address_space(1))) v4i_*)(void*)(p))
#define LAS(p) ((__attribute__((address_space(3))) v4i_*)(void*)(p))
#endif

__device__ __forceinline__ unsigned short f2bf(float f) {
  union { float f; unsigned int u; } c; c.f = f;
  unsigned int u = c.u;
  u += 0x7fffu + ((u >> 16) & 1u);       // round-to-nearest-even
  return (unsigned short)(u >> 16);
}

// ---------------- fast bf16 WMMA GEMM (full tiles only) ----------------
// C(MxN) = alpha*A(MxK)*B(KxN) [+bias[n]] [+res]; A row-major bf16.
// BNN==0: B given transposed row-major (N x K). BNN==1: B row-major (K x N).
// Requires: M % 128 == 0, N % 128 == 0, K % 32 == 0, lda/ldb multiples of 8.
// Block tile 128x128x32; 8 waves as 4(M) x 2(N); wave tile 32x64 = 8 WMMA.
#define BM 128
#define BN 128
#define BK 32
#define LDT (BK + 8)   // 40 halfs: 16B-aligned 8-half groups, offset bank conflicts

template<int BNN>
__global__ __launch_bounds__(256)
void gemm_fast(const unsigned short* __restrict__ A, int lda, long long sAz,
               const unsigned short* __restrict__ Bm, int ldb, long long sBz,
               float* __restrict__ Cf, unsigned short* __restrict__ Cb, int ldc, long long sCz,
               const float* __restrict__ bias,
               const float* __restrict__ res, int ldr, long long sRz,
               int K, float alpha)
{
  __shared__ unsigned short As[2 * BM * LDT];
  __shared__ unsigned short Bs[2 * BN * LDT];

  const int tid  = threadIdx.x;
  const int lane = tid & 31;
  const int wv   = tid >> 5;          // wave 0..7
  const int wm   = (wv & 3) * 32;     // wave M offset (4 waves along M)
  const int wn   = (wv >> 2) * 64;    // wave N offset (2 waves along N)
  const int lm   = lane & 15;
  const int lh   = lane >> 4;

  const long long z = blockIdx.z;
  const int m0 = blockIdx.y * BM;
  const int n0 = blockIdx.x * BN;

  const unsigned short* Ab = A + z * sAz + (long long)m0 * lda;
  const unsigned short* Bb = Bm + z * sBz;
  if (Cf)  Cf  += z * sCz;
  if (Cb)  Cb  += z * sCz;
  if (res) res += z * sRz;

  // A staging: 128x32 halfs = 512 8-half chunks -> 2 per thread (rows rA, rA+64)
  const int rA  = tid >> 2;
  const int kcA = (tid & 3) * 8;
  const unsigned short* pA0 = Ab + (long long)rA * lda + kcA;
  const unsigned short* pA1 = Ab + (long long)(rA + 64) * lda + kcA;

  // B staging: 128x32 halfs = 512 chunks -> 2 per thread
  int nB = 0, kcB = 0, kB = 0, ncB = 0;
  const unsigned short* pB0 = nullptr;
  const unsigned short* pB1 = nullptr;
  if (BNN == 0) {            // BT rows (n) are K-contiguous
    nB = tid >> 2; kcB = (tid & 3) * 8;
    pB0 = Bb + (long long)(n0 + nB) * ldb + kcB;
    pB1 = Bb + (long long)(n0 + nB + 64) * ldb + kcB;
  } else {                   // B rows (k) are N-contiguous; transpose on commit
    kB = tid >> 4; ncB = (tid & 15) * 8;          // kB 0..15, rows kB and kB+16
    pB0 = Bb + (long long)kB * ldb + n0 + ncB;
    pB1 = Bb + (long long)(kB + 16) * ldb + n0 + ncB;
  }

  v8f acc[2][4];
  const v8f vzero = {0.f,0.f,0.f,0.f,0.f,0.f,0.f,0.f};
  for (int i = 0; i < 2; ++i)
    for (int j = 0; j < 4; ++j) acc[i][j] = vzero;

  // WMMA compute on LDS buffer pp (ISA wave32 fragment layouts)
  auto compute = [&](int pp) {
    union Frag { uint4 q[2]; v16bf v; };
    Frag af[2], bf2[4];
    const unsigned short* as = &As[pp * BM * LDT];
    const unsigned short* bs = &Bs[pp * BN * LDT];
    for (int mt = 0; mt < 2; ++mt) {
      // A 16x32: lanes0-15 = K[0..7],K[16..23]; lanes16-31 = K[8..15],K[24..31]
      const unsigned short* p = &as[(wm + mt * 16 + lm) * LDT + lh * 8];
      af[mt].q[0] = *(const uint4*)p;
      af[mt].q[1] = *(const uint4*)(p + 16);
    }
    for (int nt = 0; nt < 4; ++nt) {
      // B 32x16: lanes0-15 = K[0..7],K[16..23]; lanes16-31 = K[8..15],K[24..31] (col = lane%16)
      const unsigned short* p = &bs[(wn + nt * 16 + lm) * LDT + lh * 8];
      bf2[nt].q[0] = *(const uint4*)p;
      bf2[nt].q[1] = *(const uint4*)(p + 16);
    }
    for (int mt = 0; mt < 2; ++mt)
      for (int nt = 0; nt < 4; ++nt)
        acc[mt][nt] = __builtin_amdgcn_wmma_f32_16x16x32_bf16(
            false, af[mt].v, false, bf2[nt].v, (short)0, acc[mt][nt], false, false);
  };

#if HAVE_ASYNC
  // ---- async global->LDS pipeline (ASYNCcnt); NN B-operand stays via regs ----
  auto async_tile = [&](int kn, int pp) {
    unsigned short* as = &As[pp * BM * LDT];
    __builtin_amdgcn_global_load_async_to_lds_b128(
        GAS(pA0 + kn), LAS(&as[rA * LDT + kcA]), 0, 0);
    __builtin_amdgcn_global_load_async_to_lds_b128(
        GAS(pA1 + kn), LAS(&as[(rA + 64) * LDT + kcA]), 0, 0);
    if (BNN == 0) {
      unsigned short* bs = &Bs[pp * BN * LDT];
      __builtin_amdgcn_global_load_async_to_lds_b128(
          GAS(pB0 + kn), LAS(&bs[nB * LDT + kcB]), 0, 0);
      __builtin_amdgcn_global_load_async_to_lds_b128(
          GAS(pB1 + kn), LAS(&bs[(nB + 64) * LDT + kcB]), 0, 0);
    }
  };
  auto commitB = [&](int pp, const uint4& b0, const uint4& b1) {
    unsigned short* bs = &Bs[pp * BN * LDT];
    union { uint4 q; unsigned short h[8]; } t0, t1; t0.q = b0; t1.q = b1;
    for (int j = 0; j < 8; ++j) {
      bs[(ncB + j) * LDT + kB]      = t0.h[j];
      bs[(ncB + j) * LDT + kB + 16] = t1.h[j];
    }
  };

  uint4 fb0, fb1;
  async_tile(0, 0);
  if (BNN == 1) {
    fb0 = *(const uint4*)pB0;
    fb1 = *(const uint4*)pB1;
    commitB(0, fb0, fb1);
  }
  __builtin_amdgcn_s_wait_asynccnt(0);
  __syncthreads();

  int pp = 0;
  for (int k0 = 0; k0 < K; k0 += BK) {
    const bool more = (k0 + BK) < K;
    if (more) {
      const int kn = k0 + BK;
      async_tile(kn, pp ^ 1);
      if (BNN == 1) {
        fb0 = *(const uint4*)(pB0 + (long long)kn * ldb);
        fb1 = *(const uint4*)(pB1 + (long long)kn * ldb);
      }
      if (k0 + 2 * BK < K) __builtin_prefetch(pA0 + k0 + 2 * BK, 0, 1);
    }
    compute(pp);
    if (more && BNN == 1) commitB(pp ^ 1, fb0, fb1);
    __builtin_amdgcn_s_wait_asynccnt(0);
    __syncthreads();
    pp ^= 1;
  }
#else
  // ---- fallback: register-staged double-buffered pipeline ----
  auto commit = [&](int pp, const uint4& a0, const uint4& a1,
                    const uint4& b0, const uint4& b1) {
    unsigned short* as = &As[pp * BM * LDT];
    *(uint4*)&as[rA * LDT + kcA]        = a0;
    *(uint4*)&as[(rA + 64) * LDT + kcA] = a1;
    unsigned short* bs = &Bs[pp * BN * LDT];
    if (BNN == 0) {
      *(uint4*)&bs[nB * LDT + kcB]        = b0;
      *(uint4*)&bs[(nB + 64) * LDT + kcB] = b1;
    } else {
      union { uint4 q; unsigned short h[8]; } t0, t1; t0.q = b0; t1.q = b1;
      for (int j = 0; j < 8; ++j) {
        bs[(ncB + j) * LDT + kB]      = t0.h[j];
        bs[(ncB + j) * LDT + kB + 16] = t1.h[j];
      }
    }
  };
  uint4 fa0 = *(const uint4*)pA0;
  uint4 fa1 = *(const uint4*)pA1;
  uint4 fb0 = *(const uint4*)pB0;
  uint4 fb1 = *(const uint4*)pB1;
  commit(0, fa0, fa1, fb0, fb1);
  __syncthreads();

  int pp = 0;
  for (int k0 = 0; k0 < K; k0 += BK) {
    const bool more = (k0 + BK) < K;
    if (more) {
      const int kn = k0 + BK;
      fa0 = *(const uint4*)(pA0 + kn);
      fa1 = *(const uint4*)(pA1 + kn);
      if (BNN == 0) {
        fb0 = *(const uint4*)(pB0 + kn);
        fb1 = *(const uint4*)(pB1 + kn);
      } else {
        fb0 = *(const uint4*)(pB0 + (long long)kn * ldb);
        fb1 = *(const uint4*)(pB1 + (long long)kn * ldb);
      }
      if (k0 + 2 * BK < K) __builtin_prefetch(pA0 + k0 + 2 * BK, 0, 1);
    }
    compute(pp);
    if (more) commit(pp ^ 1, fa0, fa1, fb0, fb1);
    __syncthreads();
    pp ^= 1;
  }
#endif

  // epilogue: unguarded (buffers padded). D: VGPR v -> M = v + 8*lh, N = lane%16
  for (int mt = 0; mt < 2; ++mt)
    for (int nt = 0; nt < 4; ++nt) {
      int n = n0 + wn + nt * 16 + lm;
      float bv = bias ? bias[n] : 0.0f;
      for (int v = 0; v < 8; ++v) {
        int m = m0 + wm + mt * 16 + lh * 8 + v;
        float val = acc[mt][nt][v] * alpha + bv;
        if (res) val += res[(long long)m * ldr + n];
        long long off = (long long)m * ldc + n;
        if (Cf) Cf[off] = val;
        if (Cb) Cb[off] = f2bf(val);
      }
    }
}

// ---------------- patchify + LayerNorm(768) -> bf16 (padded row layout) --------
__global__ __launch_bounds__(256)
void patchify_ln(const float* __restrict__ img, const float* __restrict__ g,
                 const float* __restrict__ bt, unsigned short* __restrict__ out)
{
  const int vrow = blockIdx.x;           // 0..ROWS-1 (valid rows only)
  const int bi   = vrow / NPAT;
  const int p    = vrow % NPAT;
  const long long orow = (long long)bi * TP + p;
  const int ph = p / 14, pw = p % 14;
  const int tid = threadIdx.x;
  float vals[3], s = 0.f, s2 = 0.f;
  for (int i = 0; i < 3; ++i) {
    int d  = tid + i * 256;              // patch-dim order (p1, p2, c)
    int p1 = d / 48, rem = d % 48;
    int p2 = rem / 3, c = rem % 3;
    float v = img[(((long long)bi * 3 + c) * 224 + (ph * 16 + p1)) * 224 + (pw * 16 + p2)];
    vals[i] = v; s += v; s2 += v * v;
  }
  __shared__ float r1[256], r2[256];
  r1[tid] = s; r2[tid] = s2; __syncthreads();
  for (int off = 128; off > 0; off >>= 1) {
    if (tid < off) { r1[tid] += r1[tid + off]; r2[tid] += r2[tid + off]; }
    __syncthreads();
  }
  float mean = r1[0] / 768.f;
  float var  = r2[0] / 768.f - mean * mean;
  float rs   = rsqrtf(var + EPSLN);
  for (int i = 0; i < 3; ++i) {
    int d = tid + i * 256;
    out[orow * 768 + d] = f2bf((vals[i] - mean) * rs * g[d] + bt[d]);
  }
}

// -------- LayerNorm(1024) [+pos] -> f32 (+bf16); zero-fills padded token rows ----
__global__ __launch_bounds__(256)
void ln_row(const float* __restrict__ in, const float* __restrict__ g,
            const float* __restrict__ bt, const float* __restrict__ pos,
            float* __restrict__ outf, unsigned short* __restrict__ outb)
{
  const int row = blockIdx.x;            // 0..PROWS-1
  const int tok = row & (TP - 1);
  const int tid = threadIdx.x;
  if (tok >= NPAT) {                     // padded row -> exact zeros
    for (int i = 0; i < 4; ++i) {
      int d = tid + i * 256;
      outf[(long long)row * BDIM + d] = 0.f;
      if (outb) outb[(long long)row * BDIM + d] = 0;
    }
    return;
  }
  const float* x = in + (long long)row * BDIM;
  float v[4], s = 0.f, s2 = 0.f;
  for (int i = 0; i < 4; ++i) { float t = x[tid + i * 256]; v[i] = t; s += t; s2 += t * t; }
  __shared__ float r1[256], r2[256];
  r1[tid] = s; r2[tid] = s2; __syncthreads();
  for (int off = 128; off > 0; off >>= 1) {
    if (tid < off) { r1[tid] += r1[tid + off]; r2[tid] += r2[tid + off]; }
    __syncthreads();
  }
  float mean = r1[0] / 1024.f;
  float var  = r2[0] / 1024.f - mean * mean;
  float rs   = rsqrtf(var + EPSLN);
  for (int i = 0; i < 4; ++i) {
    int d = tid + i * 256;
    float y = (v[i] - mean) * rs * g[d] + bt[d];
    if (pos) y += pos[(long long)tok * BDIM + d];
    outf[(long long)row * BDIM + d] = y;
    if (outb) outb[(long long)row * BDIM + d] = f2bf(y);
  }
}

// ------- softmax over valid 196 cols; zero-fills all padding (wave32 per row) ----
__global__ __launch_bounds__(256)
void softmax_rows(const float* __restrict__ S, unsigned short* __restrict__ P)
{
  int wid  = blockIdx.x * 8 + (threadIdx.x >> 5);   // 0..PROWS-1
  int lane = threadIdx.x & 31;
  int tok  = wid & (TP - 1);
  const float* r = S + (long long)wid * TP;
  unsigned short* o = P + (long long)wid * TP;
  if (tok >= NPAT) {
    for (int j = lane; j < TP; j += 32) o[j] = 0;
    return;
  }
  float mx = -3.4e38f;
  for (int j = lane; j < NPAT; j += 32) mx = fmaxf(mx, r[j]);
  for (int off = 16; off; off >>= 1) mx = fmaxf(mx, __shfl_xor(mx, off, 32));
  float sum = 0.f;
  for (int j = lane; j < NPAT; j += 32) sum += __expf(r[j] - mx);
  for (int off = 16; off; off >>= 1) sum += __shfl_xor(sum, off, 32);
  float inv = 1.0f / sum;
  for (int j = lane; j < TP; j += 32)
    o[j] = (j < NPAT) ? f2bf(__expf(r[j] - mx) * inv) : (unsigned short)0;
}

// -------- mean pool over valid tokens -> bf16 (rows padded to 128, zero) --------
__global__ __launch_bounds__(256)
void meanpool(const float* __restrict__ x, unsigned short* __restrict__ pb)
{
  int b = blockIdx.x, tid = threadIdx.x;   // grid = 128 (rows 64..127 zero pad)
  for (int d = tid; d < BDIM; d += 256) {
    float s = 0.f;
    if (b < NBATCH) {
      const float* p = x + ((long long)b * TP) * BDIM + d;
      for (int t = 0; t < NPAT; ++t) s += p[(long long)t * BDIM];
      s *= (1.0f / NPAT);
    }
    pb[(long long)b * BDIM + d] = f2bf(s);
  }
}

// ---- transpose + f32->bf16: W(KxN) -> WT(Npad x K), rows n>=Nd zero-filled ----
__global__ void transpose_cvt(const float* __restrict__ Wm, unsigned short* __restrict__ WT,
                              int Kd, int Nd, int Npad)
{
  long long idx = (long long)blockIdx.x * blockDim.x + threadIdx.x;
  if (idx >= (long long)Npad * Kd) return;
  int n = (int)(idx / Kd), k = (int)(idx % Kd);
  WT[idx] = (n < Nd) ? f2bf(Wm[(long long)k * Nd + n]) : (unsigned short)0;
}

// ---------------- pad classifier bias to 1024 ----------------
__global__ void pad_bias(const float* __restrict__ b, float* __restrict__ bp)
{
  int i = blockIdx.x * 256 + threadIdx.x;
  if (i < BDIM) bp[i] = (i < NCLS) ? b[i] : 0.f;
}

// ---------------- extract valid 64x1000 block from padded 128x1024 ----------------
__global__ void copy_out(const float* __restrict__ src, float* __restrict__ dst)
{
  int idx = blockIdx.x * 256 + threadIdx.x;
  if (idx >= NBATCH * NCLS) return;
  int b = idx / NCLS, j = idx % NCLS;
  dst[idx] = src[(long long)b * BDIM + j];
}

// ---------------- orchestration ----------------
extern "C" void kernel_launch(void* const* d_in, const int* in_sizes, int n_in,
                              void* d_out, int out_size, void* d_ws, size_t ws_size,
                              hipStream_t stream)
{
  (void)in_sizes; (void)n_in; (void)out_size; (void)ws_size;
  const float* image  = (const float*)d_in[0];
  const float* pos    = (const float*)d_in[1];
  const float* ln_p_g = (const float*)d_in[2];
  const float* ln_p_b = (const float*)d_in[3];
  const float* W_emb  = (const float*)d_in[4];
  const float* b_emb  = (const float*)d_in[5];
  const float* ln_e_g = (const float*)d_in[6];
  const float* ln_e_b = (const float*)d_in[7];
  const float* WV     = (const float*)d_in[8];
  const float* WK     = (const float*)d_in[9];
  const float* WQ     = (const float*)d_in[10];
  const float* ln_g   = (const float*)d_in[11];
  const float* ln_b   = (const float*)d_in[12];
  const float* W_last = (const float*)d_in[13];
  const float* b_last = (const float*)d_in[14];
  float* out = (float*)d_out;

  // ---- carve workspace (token dim padded to TP=256, classifier padded) ----
  char* base = (char*)d_ws; size_t cur = 0;
  auto carve = [&](size_t bytes) -> char* {
    char* p = base + cur; cur = (cur + bytes + 255) & ~(size_t)255; return p;
  };
  unsigned short* xp_bf   = (unsigned short*)carve((size_t)PROWS * PDIM * 2);
  float*          x       = (float*)carve((size_t)PROWS * BDIM * 4);
  float*          xn      = (float*)carve((size_t)PROWS * BDIM * 4);
  unsigned short* xn_bf   = (unsigned short*)carve((size_t)PROWS * BDIM * 2);
  unsigned short* q_bf    = (unsigned short*)carve((size_t)PROWS * BDIM * 2);
  unsigned short* k_bf    = (unsigned short*)carve((size_t)PROWS * BDIM * 2);
  unsigned short* fw_bf   = (unsigned short*)carve((size_t)PROWS * BDIM * 2);
  float*          S       = (float*)carve((size_t)NBATCH * TP * TP * 4);
  unsigned short* a_bf    = (unsigned short*)carve((size_t)NBATCH * TP * TP * 2);
  unsigned short* pool_bf = (unsigned short*)carve((size_t)128 * BDIM * 2);
  float*          out_pad = (float*)carve((size_t)128 * BDIM * 4);
  float*          bias_pad= (float*)carve((size_t)BDIM * 4);
  unsigned short* WembT   = (unsigned short*)carve((size_t)PDIM * BDIM * 2);
  unsigned short* WQT     = (unsigned short*)carve((size_t)BDIM * BDIM * 2);
  unsigned short* WKT     = (unsigned short*)carve((size_t)BDIM * BDIM * 2);
  unsigned short* WVT     = (unsigned short*)carve((size_t)VD * VD * 2);
  unsigned short* WlastT  = (unsigned short*)carve((size_t)BDIM * BDIM * 2); // padded

  // ---- weight prep ----
  { long long n;
    n = (long long)BDIM * PDIM;
    transpose_cvt<<<(unsigned)((n + 255) / 256), 256, 0, stream>>>(W_emb, WembT, PDIM, BDIM, BDIM);
    n = (long long)BDIM * BDIM;
    transpose_cvt<<<(unsigned)((n + 255) / 256), 256, 0, stream>>>(WQ, WQT, BDIM, BDIM, BDIM);
    transpose_cvt<<<(unsigned)((n + 255) / 256), 256, 0, stream>>>(WK, WKT, BDIM, BDIM, BDIM);
    n = (long long)VD * VD;
    transpose_cvt<<<(unsigned)((n + 255) / 256), 256, 0, stream>>>(WV, WVT, VD, VD, VD);
    n = (long long)BDIM * BDIM;   // pad 1000 -> 1024 output cols with zero WT rows
    transpose_cvt<<<(unsigned)((n + 255) / 256), 256, 0, stream>>>(W_last, WlastT, BDIM, NCLS, BDIM);
    pad_bias<<<4, 256, 0, stream>>>(b_last, bias_pad);
  }

  // ---- patchify + LN(768) ----
  patchify_ln<<<ROWS, 256, 0, stream>>>(image, ln_p_g, ln_p_b, xp_bf);

  // ---- embed: xn = xp @ W_emb + b_emb ----
  gemm_fast<0><<<dim3(BDIM / BN, PROWS / BM, 1), 256, 0, stream>>>(
      xp_bf, PDIM, 0, WembT, PDIM, 0,
      xn, nullptr, BDIM, 0, b_emb, nullptr, 0, 0, PDIM, 1.0f);
  // ---- x = LN_e(xn) + pos ----
  ln_row<<<PROWS, 256, 0, stream>>>(xn, ln_e_g, ln_e_b, pos, x, nullptr);

  const float scale = 0.08838834764831845f;  // 1/sqrt(128)
  for (int layer = 0; layer < NDEPTH; ++layer) {
    ln_row<<<PROWS, 256, 0, stream>>>(x, ln_g, ln_b, nullptr, xn, xn_bf);
    // Q / K projections
    gemm_fast<0><<<dim3(BDIM / BN, PROWS / BM, 1), 256, 0, stream>>>(
        xn_bf, BDIM, 0, WQT, BDIM, 0,
        nullptr, q_bf, BDIM, 0, nullptr, nullptr, 0, 0, BDIM, 1.0f);
    gemm_fast<0><<<dim3(BDIM / BN, PROWS / BM, 1), 256, 0, stream>>>(
        xn_bf, BDIM, 0, WKT, BDIM, 0,
        nullptr, k_bf, BDIM, 0, nullptr, nullptr, 0, 0, BDIM, 1.0f);
    // S = scale * Q K^T per batch (K rows K-contiguous -> TN form)
    gemm_fast<0><<<dim3(TP / BN, TP / BM, NBATCH), 256, 0, stream>>>(
        q_bf, BDIM, (long long)TP * BDIM, k_bf, BDIM, (long long)TP * BDIM,
        S, nullptr, TP, (long long)TP * TP, nullptr, nullptr, 0, 0, BDIM, scale);
    // A = softmax(S) -> bf16 (zero padding)
    softmax_rows<<<PROWS / 8, 256, 0, stream>>>(S, a_bf);
    // fw = per-head xn @ WV (gridDim.z = head, 128-column slices)
    gemm_fast<0><<<dim3(VD / BN, PROWS / BM, HEADS), 256, 0, stream>>>(
        xn_bf, BDIM, (long long)VD, WVT, VD, 0,
        nullptr, fw_bf, BDIM, (long long)VD, nullptr, nullptr, 0, 0, VD, 1.0f);
    // x = TAU*(A @ fw) + xn  (fw NN layout; K = TP padded; zeros kill pad rows)
    gemm_fast<1><<<dim3(BDIM / BN, TP / BM, NBATCH), 256, 0, stream>>>(
        a_bf, TP, (long long)TP * TP, fw_bf, BDIM, (long long)TP * BDIM,
        x, nullptr, BDIM, (long long)TP * BDIM, nullptr,
        xn, BDIM, (long long)TP * BDIM, TP, 1.0f /* TAU */);
  }

  // ---- mean pool (padded to 128 rows) + padded classifier + extract ----
  meanpool<<<128, 256, 0, stream>>>(x, pool_bf);
  gemm_fast<0><<<dim3(BDIM / BN, 1, 1), 256, 0, stream>>>(
      pool_bf, BDIM, 0, WlastT, BDIM, 0,
      out_pad, nullptr, BDIM, 0, bias_pad, nullptr, 0, 0, BDIM, 1.0f);
  copy_out<<<(NBATCH * NCLS + 255) / 256, 256, 0, stream>>>(out_pad, out);
}
